// DeepSeekAttention_40080634806364
// MI455X (gfx1250) — matrix operand order repaired and checked by reading, not compiled, not run on hardware
//
#include <hip/hip_runtime.h>
#include <hip/hip_bf16.h>

#define HIDDEN 1536
#define NH 12
#define NKV 2
#define HD 128
#define GRP 6
#define SEQ 2048
#define BATCH 2
#define MTOT (BATCH * SEQ)

typedef __attribute__((ext_vector_type(16))) __bf16 v16bf;
typedef __attribute__((ext_vector_type(8)))  float  v8f;

union AFrag { v16bf v; uint4 u[2]; };

__device__ inline v8f vzero8() {
  v8f z;
  #pragma unroll
  for (int i = 0; i < 8; ++i) z[i] = 0.f;
  return z;
}

__device__ inline v8f wmma_bf16(v16bf a, v16bf b, v8f c) {
  return __builtin_amdgcn_wmma_f32_16x16x32_bf16(false, a, false, b, (short)0, c,
                                                 false, false);
}

// Async DMA: 16B per lane, global -> LDS, tracked by ASYNCcnt (no VGPR landing).
__device__ inline void async_copy16(const void* gbase, unsigned gbyteoff,
                                    void* ldsdst) {
  unsigned lds_addr = (unsigned)(unsigned long long)ldsdst;  // LDS aperture: low 32b = LDS offset
  asm volatile("global_load_async_to_lds_b128 %0, %1, %2 offset:0"
               :: "v"(lds_addr), "v"(gbyteoff),
                  "s"((unsigned long long)gbase)
               : "memory");
}

// ---------------------------------------------------------------- f32 -> bf16
__global__ void __launch_bounds__(256)
cvt_f32_bf16(const float* __restrict__ in, __bf16* __restrict__ out, int n) {
  int i = blockIdx.x * blockDim.x + threadIdx.x;
  int stride = gridDim.x * blockDim.x;
  for (; i < n; i += stride) out[i] = (__bf16)in[i];
}

// ------------------------------------------------------------- WMMA GEMM
// C[m,n] = sum_k A[m,k] * W[n,k]   (A: MxK bf16 row-major, W: NxK bf16 row-major)
// MODE 0: bf16 out row-major [M,N]
// MODE 1: bf16 out "batch transposed": out[(m/SEQ)*N*SEQ + n*SEQ + (m%SEQ)] (V^T)
// MODE 2: f32 out row-major [M,N]
// 128 threads = 4 waves; block tile 64x64; A/B k-tiles double-buffered in LDS
// via global_load_async_to_lds_b128 (4 async instr per wave per k-step).
template <int MODE>
__global__ void __launch_bounds__(128)
wmma_gemm(const __bf16* __restrict__ A, const __bf16* __restrict__ W,
          void* __restrict__ outp, int M, int N, int K) {
  __shared__ __align__(16) __bf16 Abuf[2][64][32];
  __shared__ __align__(16) __bf16 Bbuf[2][64][32];

  const int tid  = threadIdx.x;
  const int lane = tid & 31;
  const int wv   = tid >> 5;
  const int m0   = blockIdx.y * 64;
  const int n0   = blockIdx.x * 64;
  const int lm   = lane & 15;
  const int g    = lane >> 4;

  auto prefetch = [&](int buf, int kk) {
    #pragma unroll
    for (int i = 0; i < 2; ++i) {           // 256 transfers of 16B per array
      const int t   = tid + i * 128;
      const int row = t >> 2;
      const int ch  = t & 3;
      async_copy16(A, (unsigned)(((size_t)(m0 + row) * K + kk) * 2) + ch * 16,
                   &Abuf[buf][row][ch * 8]);
      async_copy16(W, (unsigned)(((size_t)(n0 + row) * K + kk) * 2) + ch * 16,
                   &Bbuf[buf][row][ch * 8]);
    }
  };

  v8f acc[4];
  #pragma unroll
  for (int j = 0; j < 4; ++j) acc[j] = vzero8();

  const int nk = K / 32;
  prefetch(0, 0);
  for (int kt = 0; kt < nk; ++kt) {
    const int cur = kt & 1;
    if (kt + 1 < nk) {
      prefetch(cur ^ 1, (kt + 1) * 32);
      asm volatile("s_wait_asynccnt 0x4" ::: "memory");  // drain current tile only
    } else {
      asm volatile("s_wait_asynccnt 0x0" ::: "memory");
    }
    __syncthreads();

    AFrag af;
    af.u[0] = *reinterpret_cast<const uint4*>(&Abuf[cur][wv * 16 + lm][8 * g]);
    af.u[1] = *reinterpret_cast<const uint4*>(&Abuf[cur][wv * 16 + lm][16 + 8 * g]);
    #pragma unroll
    for (int j = 0; j < 4; ++j) {
      AFrag bf;
      bf.u[0] = *reinterpret_cast<const uint4*>(&Bbuf[cur][16 * j + lm][8 * g]);
      bf.u[1] = *reinterpret_cast<const uint4*>(&Bbuf[cur][16 * j + lm][16 + 8 * g]);
      acc[j] = wmma_bf16(af.v, bf.v, acc[j]);
    }
    __syncthreads();  // all waves done with `cur` before it is overwritten
  }

  #pragma unroll
  for (int j = 0; j < 4; ++j) {
    #pragma unroll
    for (int r = 0; r < 8; ++r) {
      const int m = m0 + wv * 16 + r + 8 * g;
      const int n = n0 + 16 * j + lm;
      if (MODE == 2) {
        reinterpret_cast<float*>(outp)[(size_t)m * N + n] = acc[j][r];
      } else if (MODE == 0) {
        reinterpret_cast<__bf16*>(outp)[(size_t)m * N + n] = (__bf16)acc[j][r];
      } else {
        const int b = m / SEQ, s = m % SEQ;
        reinterpret_cast<__bf16*>(outp)[((size_t)b * N + n) * SEQ + s] =
            (__bf16)acc[j][r];
      }
    }
  }
}

// ----------------------------------------------------- fused flash attention
// 4 waves/block share one (b, head); each wave owns 16 query rows and sweeps
// all keys in 32-chunks. K/V chunks staged in LDS (shared by the 4 waves) via
// async-to-LDS double buffering. Q: [B,S,NH,HD]  K: [B,S,NKV,HD]
// VT: [B,NKV,HD,S]  O: [B,S,NH*HD]  (all bf16)
__global__ void __launch_bounds__(128)
attn_fwd(const __bf16* __restrict__ Q, const __bf16* __restrict__ Kb,
         const __bf16* __restrict__ VT, const float* __restrict__ amask,
         __bf16* __restrict__ O) {
  __shared__ __align__(16) __bf16 Klds[2][32][128];  // [key][d]      16 KB
  __shared__ __align__(16) __bf16 Vlds[2][128][32];  // [d][key]      16 KB
  __shared__ __align__(16) __bf16 plds[4][16][32];   // P staging      4 KB

  const int tid  = threadIdx.x;
  const int lane = tid & 31;
  const int wv   = tid >> 5;
  const int wid0 = blockIdx.x * 4;            // all 4 waves: same b,h (4 | 128)
  const int qt   = (wid0 + wv) & (SEQ / 16 - 1);
  const int h    = (wid0 >> 7) % NH;
  const int b    = wid0 / ((SEQ / 16) * NH);
  const int kvh  = h / GRP;
  const int lm   = lane & 15;
  const int g    = lane >> 4;
  const int q0   = qt * 16;
  const float scale = 0.08838834764831845f;   // 1/sqrt(128)

  const size_t kgbase = ((size_t)b * SEQ * NKV + kvh) * HD;       // elements
  const size_t vgbase = (size_t)(b * NKV + kvh) * HD * SEQ;       // elements

  auto prefetch_kv = [&](int buf, int kb) {
    #pragma unroll
    for (int i = 0; i < 4; ++i) {             // K: 512 transfers of 16B
      const int t   = tid + i * 128;
      const int key = t >> 4;
      const int ch  = t & 15;
      async_copy16(Kb,
          (unsigned)((kgbase + (size_t)(kb + key) * (NKV * HD)) * 2) + ch * 16,
          &Klds[buf][key][ch * 8]);
    }
    #pragma unroll
    for (int i = 0; i < 4; ++i) {             // V: 512 transfers of 16B
      const int t  = tid + i * 128;
      const int dr = t >> 2;
      const int ch = t & 3;
      async_copy16(VT,
          (unsigned)((vgbase + (size_t)dr * SEQ + kb) * 2) + ch * 16,
          &Vlds[buf][dr][ch * 8]);
    }
  };

  // Q A-fragments for d = 0..127 (4 chunks of K=32), kept in registers
  v16bf qf[4];
  {
    const __bf16* qrow = Q + ((size_t)(b * SEQ + q0 + lm) * NH + h) * HD;
    #pragma unroll
    for (int dc = 0; dc < 4; ++dc) {
      AFrag f;
      f.u[0] = *reinterpret_cast<const uint4*>(qrow + 32 * dc + 8 * g);
      f.u[1] = *reinterpret_cast<const uint4*>(qrow + 32 * dc + 16 + 8 * g);
      qf[dc] = f.v;
    }
  }

  v8f oacc[8];
  #pragma unroll
  for (int j = 0; j < 8; ++j) oacc[j] = vzero8();
  float mrun[8], lrun[8];
  #pragma unroll
  for (int r = 0; r < 8; ++r) { mrun[r] = -1e30f; lrun[r] = 0.f; }

  const float* mk = amask + (size_t)b * SEQ;

  prefetch_kv(0, 0);
  for (int kt = 0; kt < SEQ / 32; ++kt) {
    const int cur = kt & 1;
    const int kb  = kt * 32;
    if (kt + 1 < SEQ / 32) {
      prefetch_kv(cur ^ 1, kb + 32);
      asm volatile("s_wait_asynccnt 0x8" ::: "memory");
    } else {
      asm volatile("s_wait_asynccnt 0x0" ::: "memory");
    }
    __syncthreads();

    // ---- scores: S[16q x 32k] = Q (16x128) x K^T (128x32), 8 WMMAs
    v8f sacc[2];
    sacc[0] = vzero8(); sacc[1] = vzero8();
    #pragma unroll
    for (int dc = 0; dc < 4; ++dc) {
      #pragma unroll
      for (int t = 0; t < 2; ++t) {
        AFrag kf;
        kf.u[0] = *reinterpret_cast<const uint4*>(
            &Klds[cur][t * 16 + lm][32 * dc + 8 * g]);
        kf.u[1] = *reinterpret_cast<const uint4*>(
            &Klds[cur][t * 16 + lm][32 * dc + 16 + 8 * g]);
        sacc[t] = wmma_bf16(qf[dc], kf.v, sacc[t]);
      }
    }

    // ---- online softmax (rows in components; keys across 16 lanes)
    const float mt0 = (1.f - mk[kb + lm]) * -10000.f;
    const float mt1 = (1.f - mk[kb + 16 + lm]) * -10000.f;
    float p0[8], p1[8], alpha[8];
    #pragma unroll
    for (int r = 0; r < 8; ++r) {
      float s0 = sacc[0][r] * scale + mt0;
      float s1 = sacc[1][r] * scale + mt1;
      float mx = fmaxf(s0, s1);
      #pragma unroll
      for (int mm = 1; mm < 16; mm <<= 1) mx = fmaxf(mx, __shfl_xor(mx, mm, 32));
      const float mn = fmaxf(mrun[r], mx);
      alpha[r] = __expf(mrun[r] - mn);
      mrun[r]  = mn;
      p0[r] = __expf(s0 - mn);
      p1[r] = __expf(s1 - mn);
      float ps = p0[r] + p1[r];
      #pragma unroll
      for (int mm = 1; mm < 16; mm <<= 1) ps += __shfl_xor(ps, mm, 32);
      lrun[r] = lrun[r] * alpha[r] + ps;
    }
    #pragma unroll
    for (int j = 0; j < 8; ++j) {
      #pragma unroll
      for (int r = 0; r < 8; ++r) oacc[j][r] *= alpha[r];
    }

    // ---- C-layout -> A-layout for P via wave-private LDS
    #pragma unroll
    for (int r = 0; r < 8; ++r) {
      plds[wv][r + 8 * g][lm]      = (__bf16)p0[r];
      plds[wv][r + 8 * g][16 + lm] = (__bf16)p1[r];
    }
    asm volatile("s_wait_dscnt 0" ::: "memory");  // same-wave LDS ordering
    AFrag pf;
    pf.u[0] = *reinterpret_cast<const uint4*>(&plds[wv][lm][8 * g]);
    pf.u[1] = *reinterpret_cast<const uint4*>(&plds[wv][lm][16 + 8 * g]);

    // ---- O += P (16x32) x V (32x128), 8 WMMAs (V^T rows contiguous in LDS)
    #pragma unroll
    for (int j = 0; j < 8; ++j) {
      AFrag vf;
      vf.u[0] = *reinterpret_cast<const uint4*>(&Vlds[cur][16 * j + lm][8 * g]);
      vf.u[1] = *reinterpret_cast<const uint4*>(&Vlds[cur][16 * j + lm][16 + 8 * g]);
      oacc[j] = wmma_bf16(pf.v, vf.v, oacc[j]);
    }
    __syncthreads();  // protect `cur` before next overwrite
  }

  // ---- epilogue: normalize and store bf16 into O[b, s, h*HD + d]
  float rinv[8];
  #pragma unroll
  for (int r = 0; r < 8; ++r) rinv[r] = 1.f / lrun[r];
  #pragma unroll
  for (int j = 0; j < 8; ++j) {
    #pragma unroll
    for (int r = 0; r < 8; ++r) {
      O[(size_t)(b * SEQ + q0 + r + 8 * g) * HIDDEN + h * HD + 16 * j + lm] =
          (__bf16)(oacc[j][r] * rinv[r]);
    }
  }
}

// ----------------------------------------------------------------- launcher
extern "C" void kernel_launch(void* const* d_in, const int* in_sizes, int n_in,
                              void* d_out, int out_size, void* d_ws, size_t ws_size,
                              hipStream_t stream) {
  const float* hs    = (const float*)d_in[0];
  const float* amask = (const float*)d_in[1];
  const float* Wq    = (const float*)d_in[2];
  const float* Wk    = (const float*)d_in[3];
  const float* Wv    = (const float*)d_in[4];
  const float* Wo    = (const float*)d_in[5];

  char* ws = (char*)d_ws;
  size_t off = 0;
  auto alloc = [&](size_t elems) {
    __bf16* p = (__bf16*)(ws + off);
    off += elems * sizeof(__bf16);
    return p;
  };
  __bf16* Xbf  = alloc((size_t)MTOT * HIDDEN);        // 12.6 MB
  __bf16* Wqbf = alloc((size_t)HIDDEN * HIDDEN);      // 4.7 MB
  __bf16* Wkbf = alloc((size_t)NKV * HD * HIDDEN);    // 0.8 MB
  __bf16* Wvbf = alloc((size_t)NKV * HD * HIDDEN);    // 0.8 MB
  __bf16* Wobf = alloc((size_t)HIDDEN * HIDDEN);      // 4.7 MB
  __bf16* Qbf  = alloc((size_t)MTOT * HIDDEN);        // 12.6 MB [B,S,NH,HD]
  __bf16* Kbf  = alloc((size_t)MTOT * NKV * HD);      // 2.1 MB  [B,S,NKV,HD]
  __bf16* VTbf = alloc((size_t)MTOT * NKV * HD);      // 2.1 MB  [B,NKV,HD,S]
  __bf16* Obf  = alloc((size_t)MTOT * HIDDEN);        // 12.6 MB [B,S,NH*HD]

  cvt_f32_bf16<<<2048, 256, 0, stream>>>(hs, Xbf, MTOT * HIDDEN);
  cvt_f32_bf16<<<2048, 256, 0, stream>>>(Wq, Wqbf, HIDDEN * HIDDEN);
  cvt_f32_bf16<<<512, 256, 0, stream>>>(Wk, Wkbf, NKV * HD * HIDDEN);
  cvt_f32_bf16<<<512, 256, 0, stream>>>(Wv, Wvbf, NKV * HD * HIDDEN);
  cvt_f32_bf16<<<2048, 256, 0, stream>>>(Wo, Wobf, HIDDEN * HIDDEN);

  dim3 blk(128);
  // Q = X Wq^T  -> [B,S,NH,HD] bf16
  wmma_gemm<0><<<dim3(HIDDEN / 64, MTOT / 64), blk, 0, stream>>>(
      Xbf, Wqbf, Qbf, MTOT, HIDDEN, HIDDEN);
  // K = X Wk^T  -> [B,S,NKV,HD] bf16
  wmma_gemm<0><<<dim3((NKV * HD) / 64, MTOT / 64), blk, 0, stream>>>(
      Xbf, Wkbf, Kbf, MTOT, NKV * HD, HIDDEN);
  // V = X Wv^T  -> transposed epilogue [B,NKV,HD,S] bf16
  wmma_gemm<1><<<dim3((NKV * HD) / 64, MTOT / 64), blk, 0, stream>>>(
      Xbf, Wvbf, VTbf, MTOT, NKV * HD, HIDDEN);
  // fused softmax(QK^T)V  -> [B,S,HIDDEN] bf16
  attn_fwd<<<(BATCH * NH * (SEQ / 16)) / 4, 128, 0, stream>>>(
      Qbf, Kbf, VTbf, amask, Obf);
  // out = O Wo^T -> f32 d_out
  wmma_gemm<2><<<dim3(HIDDEN / 64, MTOT / 64), blk, 0, stream>>>(
      Obf, Wobf, d_out, MTOT, HIDDEN, HIDDEN);
}
